// SampledArcFaceLayer_21114059227271
// MI455X (gfx1250) — compile-verified, matrix operand order checked by reading, not run on hardware
//
#include <hip/hip_runtime.h>

// ---------------- problem constants (match reference) ----------------
#define B_ROWS     8192
#define DIM        512
#define N_SMP      5000
#define N_PAD      5120          // padded N: 40 tiles of 128

#define ARC_SCALE  30.0f
#define ARC_COS_M  0.87758256189037276f   // cos(0.5)
#define ARC_SIN_M  0.47942553860420301f   // sin(0.5)
#define ARC_TH    -0.87758256189037276f   // cos(pi-0.5)
#define ARC_MM     0.23971276930210150f   // sin(pi-0.5)*0.5

// ---------------- vector types ----------------
typedef __attribute__((ext_vector_type(16))) __bf16 bf16x16;
typedef __attribute__((ext_vector_type(8)))  __bf16 bf16x8;
typedef __attribute__((ext_vector_type(8)))  float  f32x8;

union FragU {
    bf16x16 v;
    bf16x8  h[2];
};

// ---------------- kernel 1/2: row-normalize + bf16 hi/lo split ----------------
// One 256-thread block per row; each thread owns 2 of the 512 columns.
// Rows >= valid_rows (weight padding) are zero-filled.
__global__ __launch_bounds__(256) void normalize_split_kernel(
    const float* __restrict__ src, __bf16* __restrict__ hi,
    __bf16* __restrict__ lo, int valid_rows)
{
    const int row = blockIdx.x;
    const int tid = threadIdx.x;
    const size_t base = (size_t)row * DIM;

    if (row >= valid_rows) {
        hi[base + tid]       = (__bf16)0.0f;
        hi[base + tid + 256] = (__bf16)0.0f;
        lo[base + tid]       = (__bf16)0.0f;
        lo[base + tid + 256] = (__bf16)0.0f;
        return;
    }

    const float x0 = src[base + tid];
    const float x1 = src[base + tid + 256];

    __shared__ float red[256];
    red[tid] = x0 * x0 + x1 * x1;
    __syncthreads();
    #pragma unroll
    for (int s = 128; s > 0; s >>= 1) {
        if (tid < s) red[tid] += red[tid + s];
        __syncthreads();
    }
    const float rn = 1.0f / sqrtf(red[0]);

    const float n0 = x0 * rn;
    const float n1 = x1 * rn;
    const __bf16 h0 = (__bf16)n0;
    const __bf16 h1 = (__bf16)n1;
    hi[base + tid]       = h0;
    hi[base + tid + 256] = h1;
    lo[base + tid]       = (__bf16)(n0 - (float)h0);
    lo[base + tid + 256] = (__bf16)(n1 - (float)h1);
}

// ---------------- kernel 3: label remap into d_out tail ----------------
__global__ __launch_bounds__(256) void remap_kernel(
    const int* __restrict__ labels, const int* __restrict__ class_map,
    int* __restrict__ remapped)
{
    const int i = blockIdx.x * blockDim.x + threadIdx.x;
    if (i < B_ROWS) remapped[i] = class_map[labels[i]];
}

// ---------------- kernel 4: WMMA GEMM + ArcFace epilogue ----------------
// Block: 256 threads = 8 wave32s, block tile 128(M) x 128(N).
// Waves: 2(M) x 4(N); wave tile 64(M) x 32(N) = 4x2 accumulators of 16x16.
// K loop: steps of 32, 3 bf16 WMMAs per tile-pair (hi*hi + hi*lo + lo*hi).
__global__ __launch_bounds__(256) void arcface_gemm_kernel(
    const __bf16* __restrict__ Ahi, const __bf16* __restrict__ Alo,
    const __bf16* __restrict__ Bhi, const __bf16* __restrict__ Blo,
    const int*    __restrict__ remapped, float* __restrict__ out)
{
    const int lane = threadIdx.x & 31;
    const int wave = threadIdx.x >> 5;
    const int wm   = wave >> 2;          // 0..1
    const int wn   = wave & 3;           // 0..3

    const int tileM = blockIdx.y * 128 + wm * 64;
    const int tileN = blockIdx.x * 128 + wn * 32;

    const int lmod = lane & 15;          // row (A) / col (B) within 16
    const int kofs = (lane >> 4) * 8;    // lanes 16-31 start at K=8

    f32x8 acc[4][2];
    #pragma unroll
    for (int i = 0; i < 4; ++i)
        #pragma unroll
        for (int j = 0; j < 2; ++j)
            acc[i][j] = (f32x8)0.0f;

    for (int k = 0; k < DIM; k += 32) {
        FragU ah[4], al[4], bh[2], bl[2];

        #pragma unroll
        for (int t = 0; t < 4; ++t) {
            const size_t off = (size_t)(tileM + t * 16 + lmod) * DIM + k + kofs;
            ah[t].h[0] = *(const bf16x8*)(Ahi + off);
            ah[t].h[1] = *(const bf16x8*)(Ahi + off + 16);
            al[t].h[0] = *(const bf16x8*)(Alo + off);
            al[t].h[1] = *(const bf16x8*)(Alo + off + 16);
        }
        #pragma unroll
        for (int t = 0; t < 2; ++t) {
            const size_t off = (size_t)(tileN + t * 16 + lmod) * DIM + k + kofs;
            bh[t].h[0] = *(const bf16x8*)(Bhi + off);
            bh[t].h[1] = *(const bf16x8*)(Bhi + off + 16);
            bl[t].h[0] = *(const bf16x8*)(Blo + off);
            bl[t].h[1] = *(const bf16x8*)(Blo + off + 16);
        }

        #pragma unroll
        for (int tm = 0; tm < 4; ++tm) {
            #pragma unroll
            for (int tn = 0; tn < 2; ++tn) {
                acc[tm][tn] = __builtin_amdgcn_wmma_f32_16x16x32_bf16(
                    false, ah[tm].v, false, bh[tn].v, (short)0, acc[tm][tn], false, false);
                acc[tm][tn] = __builtin_amdgcn_wmma_f32_16x16x32_bf16(
                    false, ah[tm].v, false, bl[tn].v, (short)0, acc[tm][tn], false, false);
                acc[tm][tn] = __builtin_amdgcn_wmma_f32_16x16x32_bf16(
                    false, al[tm].v, false, bh[tn].v, (short)0, acc[tm][tn], false, false);
            }
        }
    }

    // Epilogue: C/D layout — lane L: col = L%16; VGPR v: row = v + 8*(L/16).
    const int rowHalf = (lane >> 4) * 8;
    #pragma unroll
    for (int tn = 0; tn < 2; ++tn) {
        const int col = tileN + tn * 16 + lmod;
        if (col >= N_SMP) continue;
        #pragma unroll
        for (int tm = 0; tm < 4; ++tm) {
            #pragma unroll
            for (int v = 0; v < 8; ++v) {
                const int row = tileM + tm * 16 + rowHalf + v;
                const float c = acc[tm][tn][v];
                const int lbl = remapped[row];
                const float s   = sqrtf(fmaxf(1.0f - c * c, 0.0f));
                float phi = c * ARC_COS_M - s * ARC_SIN_M;
                phi = (c > ARC_TH) ? phi : (c - ARC_MM);
                out[(size_t)row * N_SMP + col] = ((col == lbl) ? phi : c) * ARC_SCALE;
            }
        }
    }
}

// ---------------- launch ----------------
extern "C" void kernel_launch(void* const* d_in, const int* in_sizes, int n_in,
                              void* d_out, int out_size, void* d_ws, size_t ws_size,
                              hipStream_t stream) {
    const float* emb    = (const float*)d_in[0];
    const int*   labels = (const int*)d_in[1];
    const float* weight = (const float*)d_in[2];
    const int*   cmap   = (const int*)d_in[3];

    float* out = (float*)d_out;
    int* remapped = (int*)d_out + (size_t)B_ROWS * N_SMP;   // tail: second tuple output

    // workspace layout (~26 MB total)
    __bf16* ehi = (__bf16*)d_ws;
    __bf16* elo = ehi + (size_t)B_ROWS * DIM;
    __bf16* whi = elo + (size_t)B_ROWS * DIM;
    __bf16* wlo = whi + (size_t)N_PAD * DIM;

    normalize_split_kernel<<<B_ROWS, 256, 0, stream>>>(emb, ehi, elo, B_ROWS);
    normalize_split_kernel<<<N_PAD, 256, 0, stream>>>(weight, whi, wlo, N_SMP);
    remap_kernel<<<(B_ROWS + 255) / 256, 256, 0, stream>>>(labels, cmap, remapped);

    dim3 grid(N_PAD / 128, B_ROWS / 128);   // 40 x 64 blocks
    arcface_gemm_kernel<<<grid, 256, 0, stream>>>(ehi, elo, whi, wlo, remapped, out);
}
